// AttentionCritic_81570018885757
// MI455X (gfx1250) — matrix operand order repaired and checked by reading, not compiled, not run on hardware
//
#include <hip/hip_runtime.h>

typedef __attribute__((ext_vector_type(16))) __bf16 v16bf;
typedef __attribute__((ext_vector_type(8)))  float  v8f;

#define A_AGENTS 8
#define BATCH    32768
#define SDIM     128
#define ADIM     16
#define HDIM     64
#define NHEADS   4
#define DHEAD    16

constexpr int SA_STR  = 168;  // padded LDS stride (bf16 elems) for the [128,160] sa tile
constexpr int ENC_STR = 72;   // padded stride for s_enc / sa_enc tiles
constexpr int CIN_STR = 136;  // padded stride for critic_in tile

// ---- WMMA A-fragment loader (contiguous k = 16*hl + s order, matching packed B, wave32) ----
// Per lane the needed elements are two contiguous 16B runs -> compiles to 2x ds_load_b128.
__device__ __forceinline__ v16bf load_a_frag(const __bf16* __restrict__ base, int stride, int lane) {
  int hl = (lane >> 4) & 1;
  int m  = lane & 15;
  v16bf a;
#pragma unroll
  for (int v = 0; v < 8; ++v) {
    int kb = 2 * v + 16 * hl;
    a[2 * v]     = base[m * stride + kb];
    a[2 * v + 1] = base[m * stride + kb + 1];
  }
  return a;
}

// B fragments are pre-swizzled in LDS: tile (kt,nt) = 32 lanes x 16 contiguous bf16.
// packed index: ((kt*4 + nt)*32 + lane)*16 + e  with  e = k&15, lane = ((k>>4)&1)*16 + (n&15)
__device__ __forceinline__ v16bf load_b_packed(const __bf16* __restrict__ pk, int kt, int nt, int lane) {
  return *(const v16bf*)(pk + ((kt * 4 + nt) * 32 + lane) * 16);
}

// scatter one float4 (4 consecutive n, same k) of a [K,64] weight into packed-B LDS
__device__ __forceinline__ void pack_b4(__bf16* __restrict__ pk, int k, int n4, float4 v) {
  int kt = k >> 5;
  int hl = (k >> 4) & 1;
  int e  = k & 15;
  int nt = n4 >> 4;
  int nn = n4 & 15;
  __bf16* t = pk + (kt * 4 + nt) * 512 + e;
  t[(hl * 16 + nn + 0) * 16] = (__bf16)v.x;
  t[(hl * 16 + nn + 1) * 16] = (__bf16)v.y;
  t[(hl * 16 + nn + 2) * 16] = (__bf16)v.z;
  t[(hl * 16 + nn + 3) * 16] = (__bf16)v.w;
}

// copy one 16-row x 64-col bf16 stripe from LDS to global ([*,64] row-major), 128-bit chunks
__device__ __forceinline__ void store_stripe(__bf16* __restrict__ g, const __bf16* __restrict__ l,
                                             int lstride, int lane) {
#pragma unroll
  for (int t = 0; t < 4; ++t) {
    int chunk = lane + 32 * t;            // 128 chunks of 8 bf16
    int r = chunk >> 3, o = (chunk & 7) * 8;
    *(uint4*)(g + r * 64 + o) = *(const uint4*)(l + r * lstride + o);
  }
}

__device__ __forceinline__ void cvt4(__bf16* d, float4 v) {
  d[0] = (__bf16)v.x; d[1] = (__bf16)v.y; d[2] = (__bf16)v.z; d[3] = (__bf16)v.w;
}

__device__ __forceinline__ float bfbits_lo(unsigned u) {
  return __builtin_bit_cast(float, u << 16);
}
__device__ __forceinline__ float bfbits_hi(unsigned u) {
  return __builtin_bit_cast(float, u & 0xffff0000u);
}
// load 16 bf16 (32B aligned) from LDS, expand to f32 (exact: bits<<16)
__device__ __forceinline__ void load16f(const __bf16* __restrict__ p, float* f) {
  uint4 a = *(const uint4*)p;
  uint4 b = *(const uint4*)(p + 8);
  unsigned u[8] = {a.x, a.y, a.z, a.w, b.x, b.y, b.z, b.w};
#pragma unroll
  for (int t = 0; t < 8; ++t) { f[2 * t] = bfbits_lo(u[t]); f[2 * t + 1] = bfbits_hi(u[t]); }
}
__device__ __forceinline__ unsigned pack2bf(float lo, float hi) {
  unsigned a = (unsigned)__builtin_bit_cast(unsigned short, (__bf16)lo);
  unsigned b = (unsigned)__builtin_bit_cast(unsigned short, (__bf16)hi);
  return a | (b << 16);
}

// =====================================================================
// Kernel 1: encoders + per-head key/val/sel projections
// =====================================================================
__global__ __launch_bounds__(256) void k_encode(
    const float* __restrict__ states, const float* __restrict__ actions,
    const float* __restrict__ Ws, const float* __restrict__ bs,
    const float* __restrict__ We, const float* __restrict__ be,
    const float* __restrict__ Wk, const float* __restrict__ Wsel,
    const float* __restrict__ Wv, const float* __restrict__ bv,
    __bf16* __restrict__ saenc_g, __bf16* __restrict__ keys_g,
    __bf16* __restrict__ vals_g, __bf16* __restrict__ sel_g) {
  __shared__ __align__(32) __bf16 sSA[128 * SA_STR];
  __shared__ __align__(32) __bf16 sWsP[4 * 4 * 512];   // packed-B: 4 k-tiles
  __shared__ __align__(32) __bf16 sWeP[5 * 4 * 512];   // packed-B: 5 k-tiles
  __shared__ __align__(32) __bf16 sWkP[2 * 4 * 512];
  __shared__ __align__(32) __bf16 sWselP[2 * 4 * 512];
  __shared__ __align__(32) __bf16 sWvP[2 * 4 * 512];
  __shared__ __align__(32) __bf16 sSenc[128 * ENC_STR];
  __shared__ __align__(32) __bf16 sSAenc[128 * ENC_STR];
  __shared__ __align__(32) __bf16 sStage[128 * 64];
  __shared__ float sBs[64], sBe[64], sBv[64];

  const int tid = threadIdx.x;
  const int ag  = blockIdx.x >> 8;            // 256 tiles of 128 rows per agent
  const int b0  = (blockIdx.x & 255) * 128;

  // states tile -> sSA[:, 0:128]
  const float* sp = states + ((size_t)ag * BATCH + b0) * SDIM;
  for (int idx = tid; idx < 128 * 32; idx += 256) {
    int r = idx >> 5, c = (idx & 31) * 4;
    cvt4(&sSA[r * SA_STR + c], *(const float4*)(sp + r * SDIM + c));
  }
  // actions tile -> sSA[:, 128:144]
  const float* ap = actions + ((size_t)ag * BATCH + b0) * ADIM;
  for (int idx = tid; idx < 128 * 4; idx += 256) {
    int r = idx >> 2, c = (idx & 3) * 4;
    cvt4(&sSA[r * SA_STR + 128 + c], *(const float4*)(ap + r * ADIM + c));
  }
  // zero pad cols 144..167
  for (int idx = tid; idx < 128 * 6; idx += 256) {
    int r = idx / 6, c = 144 + (idx % 6) * 4;
    __bf16* d = &sSA[r * SA_STR + c];
    d[0] = d[1] = d[2] = d[3] = (__bf16)0.f;
  }
  // Ws [128,64] -> packed
  for (int idx = tid; idx < 2048; idx += 256) {
    int k = idx >> 4, n4 = (idx & 15) * 4;
    pack_b4(sWsP, k, n4, *(const float4*)(Ws + k * 64 + n4));
  }
  // We [144,64] -> packed; zero-pad k rows 144..159
  for (int idx = tid; idx < 2304; idx += 256) {
    int k = idx >> 4, n4 = (idx & 15) * 4;
    pack_b4(sWeP, k, n4, *(const float4*)(We + k * 64 + n4));
  }
  for (int idx = tid; idx < 256; idx += 256) {
    int nt = idx >> 6, rem = idx & 63, lane = 16 + (rem >> 2), e4 = (rem & 3) * 4;
    __bf16* d = &sWeP[(16 + nt) * 512 + lane * 16 + e4];  // kt=4 tiles
    d[0] = d[1] = d[2] = d[3] = (__bf16)0.f;
  }
  // Wk/Wsel/Wv: [NH,H,D]; concat column c = n*16+d; k index = h
  for (int idx = tid; idx < 1024; idx += 256) {
    int h = idx >> 4, rem = idx & 15, n = rem >> 2, d4 = (rem & 3) * 4;
    int goff = n * (HDIM * DHEAD) + h * DHEAD + d4;
    int n4 = n * 16 + d4;
    pack_b4(sWkP,   h, n4, *(const float4*)(Wk   + goff));
    pack_b4(sWselP, h, n4, *(const float4*)(Wsel + goff));
    pack_b4(sWvP,   h, n4, *(const float4*)(Wv   + goff));
  }
  if (tid < 64) { sBs[tid] = bs[tid]; sBe[tid] = be[tid]; sBv[tid] = bv[tid]; }
  __syncthreads();

  const int wave = tid >> 5, lane = tid & 31;
  const int m0 = wave * 16;
  const int hl = lane >> 4, nn = lane & 15;

  // Load all A fragments of the sa tile once (shared by GEMM1 and GEMM2)
  v16bf aSA[5];
#pragma unroll
  for (int kt = 0; kt < 5; ++kt)
    aSA[kt] = load_a_frag(&sSA[m0 * SA_STR + kt * 32], SA_STR, lane);

  // GEMM1: s_enc = relu(sa[:,0:128] @ Ws + bs)
#pragma unroll
  for (int n0 = 0; n0 < 64; n0 += 16) {
    v8f c = {0.f, 0.f, 0.f, 0.f, 0.f, 0.f, 0.f, 0.f};
#pragma unroll
    for (int kt = 0; kt < 4; ++kt) {
      v16bf bf = load_b_packed(sWsP, kt, n0 >> 4, lane);
      c = __builtin_amdgcn_wmma_f32_16x16x32_bf16(false, aSA[kt], false, bf, (short)0, c, false, false);
    }
    float bias = sBs[n0 + nn];
#pragma unroll
    for (int v = 0; v < 8; ++v) {
      int row = m0 + v + 8 * hl;
      float x = c[v] + bias;
      sSenc[row * ENC_STR + n0 + nn] = (__bf16)(x > 0.f ? x : 0.f);
    }
  }

  // GEMM2: sa_enc = relu(sa[:,0:160] @ We_pad + be)
#pragma unroll
  for (int n0 = 0; n0 < 64; n0 += 16) {
    v8f c = {0.f, 0.f, 0.f, 0.f, 0.f, 0.f, 0.f, 0.f};
#pragma unroll
    for (int kt = 0; kt < 5; ++kt) {
      v16bf bf = load_b_packed(sWeP, kt, n0 >> 4, lane);
      c = __builtin_amdgcn_wmma_f32_16x16x32_bf16(false, aSA[kt], false, bf, (short)0, c, false, false);
    }
    float bias = sBe[n0 + nn];
#pragma unroll
    for (int v = 0; v < 8; ++v) {
      int row = m0 + v + 8 * hl;
      float x = c[v] + bias;
      sSAenc[row * ENC_STR + n0 + nn] = (__bf16)(x > 0.f ? x : 0.f);
    }
  }

  const size_t gbase = ((size_t)ag * BATCH + b0 + m0) * 64;
  store_stripe(saenc_g + gbase, &sSAenc[m0 * ENC_STR], ENC_STR, lane);

  // A fragments for the projections (wave-local stripes just written)
  v16bf aE[2], aS[2];
#pragma unroll
  for (int kt = 0; kt < 2; ++kt) {
    aE[kt] = load_a_frag(&sSAenc[m0 * ENC_STR + kt * 32], ENC_STR, lane);
    aS[kt] = load_a_frag(&sSenc[m0 * ENC_STR + kt * 32], ENC_STR, lane);
  }

  // keys = sa_enc @ Wk_cat
#pragma unroll
  for (int n0 = 0; n0 < 64; n0 += 16) {
    v8f c = {0.f, 0.f, 0.f, 0.f, 0.f, 0.f, 0.f, 0.f};
#pragma unroll
    for (int kt = 0; kt < 2; ++kt) {
      v16bf bf = load_b_packed(sWkP, kt, n0 >> 4, lane);
      c = __builtin_amdgcn_wmma_f32_16x16x32_bf16(false, aE[kt], false, bf, (short)0, c, false, false);
    }
#pragma unroll
    for (int v = 0; v < 8; ++v)
      sStage[(m0 + v + 8 * hl) * 64 + n0 + nn] = (__bf16)c[v];
  }
  store_stripe(keys_g + gbase, &sStage[m0 * 64], 64, lane);

  // vals = relu(sa_enc @ Wv_cat + bv)
#pragma unroll
  for (int n0 = 0; n0 < 64; n0 += 16) {
    v8f c = {0.f, 0.f, 0.f, 0.f, 0.f, 0.f, 0.f, 0.f};
#pragma unroll
    for (int kt = 0; kt < 2; ++kt) {
      v16bf bf = load_b_packed(sWvP, kt, n0 >> 4, lane);
      c = __builtin_amdgcn_wmma_f32_16x16x32_bf16(false, aE[kt], false, bf, (short)0, c, false, false);
    }
    float bias = sBv[n0 + nn];
#pragma unroll
    for (int v = 0; v < 8; ++v) {
      float x = c[v] + bias;
      sStage[(m0 + v + 8 * hl) * 64 + n0 + nn] = (__bf16)(x > 0.f ? x : 0.f);
    }
  }
  store_stripe(vals_g + gbase, &sStage[m0 * 64], 64, lane);

  // sel = s_enc @ Wsel_cat
#pragma unroll
  for (int n0 = 0; n0 < 64; n0 += 16) {
    v8f c = {0.f, 0.f, 0.f, 0.f, 0.f, 0.f, 0.f, 0.f};
#pragma unroll
    for (int kt = 0; kt < 2; ++kt) {
      v16bf bf = load_b_packed(sWselP, kt, n0 >> 4, lane);
      c = __builtin_amdgcn_wmma_f32_16x16x32_bf16(false, aS[kt], false, bf, (short)0, c, false, false);
    }
#pragma unroll
    for (int v = 0; v < 8; ++v)
      sStage[(m0 + v + 8 * hl) * 64 + n0 + nn] = (__bf16)c[v];
  }
  store_stripe(sel_g + gbase, &sStage[m0 * 64], 64, lane);
}

// =====================================================================
// Kernel 2: exclude-self attention across agents (VALU)
// =====================================================================
__global__ __launch_bounds__(256) void k_attend(
    const __bf16* __restrict__ keys_g, const __bf16* __restrict__ sel_g,
    const __bf16* __restrict__ vals_g, __bf16* __restrict__ other_g) {
  __shared__ __align__(32) __bf16 sK[8 * 32 * 64];
  __shared__ __align__(32) __bf16 sS[8 * 32 * 64];
  __shared__ __align__(32) __bf16 sV[8 * 32 * 64];
  __shared__ __align__(32) __bf16 sO[8 * 32 * 64];
  const int tid = threadIdx.x;
  const int b0 = blockIdx.x * 32;

  for (int chunk = tid; chunk < 2048; chunk += 256) {
    int ag = chunk >> 8, local = chunk & 255, r = local >> 3, o = (local & 7) * 8;
    size_t g = ((size_t)ag * BATCH + b0 + r) * 64 + o;
    *(uint4*)(sK + chunk * 8) = *(const uint4*)(keys_g + g);
    *(uint4*)(sS + chunk * 8) = *(const uint4*)(sel_g + g);
    *(uint4*)(sV + chunk * 8) = *(const uint4*)(vals_g + g);
  }
  __syncthreads();

  for (int it = tid; it < 1024; it += 256) {
    int bl = it & 31, rest = it >> 5;
    int n = rest & 3, i = rest >> 2;
    int cb = n * 16;
    float selv[16];
    load16f(&sS[(i * 32 + bl) * 64 + cb], selv);

    float logit[8];
    float mx = -3.0e38f;
#pragma unroll
    for (int j = 0; j < 8; ++j) {
      float l;
      if (j == i) {
        l = -1e9f;
      } else {
        float kj[16];
        load16f(&sK[(j * 32 + bl) * 64 + cb], kj);
        float acc = 0.f;
#pragma unroll
        for (int d = 0; d < 16; ++d) acc += selv[d] * kj[d];
        l = acc * 0.25f;  // 1/sqrt(D), D=16
      }
      logit[j] = l;
      mx = fmaxf(mx, l);
    }
    float w[8], se = 0.f;
#pragma unroll
    for (int j = 0; j < 8; ++j) { float e = __expf(logit[j] - mx); w[j] = e; se += e; }
    float inv = 1.f / se;

    float outv[16];
#pragma unroll
    for (int d = 0; d < 16; ++d) outv[d] = 0.f;
#pragma unroll
    for (int j = 0; j < 8; ++j) {
      float vj[16];
      load16f(&sV[(j * 32 + bl) * 64 + cb], vj);
      float wj = w[j];
#pragma unroll
      for (int d = 0; d < 16; ++d) outv[d] += wj * vj[d];
    }
    unsigned p0 = pack2bf(outv[0] * inv,  outv[1] * inv);
    unsigned p1 = pack2bf(outv[2] * inv,  outv[3] * inv);
    unsigned p2 = pack2bf(outv[4] * inv,  outv[5] * inv);
    unsigned p3 = pack2bf(outv[6] * inv,  outv[7] * inv);
    unsigned p4 = pack2bf(outv[8] * inv,  outv[9] * inv);
    unsigned p5 = pack2bf(outv[10] * inv, outv[11] * inv);
    unsigned p6 = pack2bf(outv[12] * inv, outv[13] * inv);
    unsigned p7 = pack2bf(outv[14] * inv, outv[15] * inv);
    uint4 lo; lo.x = p0; lo.y = p1; lo.z = p2; lo.w = p3;
    uint4 hi; hi.x = p4; hi.y = p5; hi.z = p6; hi.w = p7;
    int off = (i * 32 + bl) * 64 + cb;
    *(uint4*)(&sO[off]) = lo;
    *(uint4*)(&sO[off + 8]) = hi;
  }
  __syncthreads();

  for (int chunk = tid; chunk < 2048; chunk += 256) {
    int ag = chunk >> 8, local = chunk & 255, r = local >> 3, o = (local & 7) * 8;
    size_t g = ((size_t)ag * BATCH + b0 + r) * 64 + o;
    *(uint4*)(other_g + g) = *(const uint4*)(sO + chunk * 8);
  }
}

// =====================================================================
// Kernel 3: critic head
// =====================================================================
__global__ __launch_bounds__(256) void k_critic(
    const __bf16* __restrict__ saenc_g, const __bf16* __restrict__ other_g,
    const float* __restrict__ W1, const float* __restrict__ b1,
    const float* __restrict__ W2, const float* __restrict__ b2,
    float* __restrict__ out) {
  __shared__ __align__(32) __bf16 sCin[128 * CIN_STR];
  __shared__ __align__(32) __bf16 sW1P[4 * 4 * 512];
  __shared__ float sB1[64], sW2[64];
  const int tid = threadIdx.x;
  const int ag = blockIdx.x >> 8;
  const int b0 = (blockIdx.x & 255) * 128;

  for (int chunk = tid; chunk < 1024; chunk += 256) {
    int r = chunk >> 3, o = (chunk & 7) * 8;
    size_t g = ((size_t)ag * BATCH + b0 + r) * 64 + o;
    *(uint4*)(&sCin[r * CIN_STR + o])      = *(const uint4*)(saenc_g + g);
    *(uint4*)(&sCin[r * CIN_STR + 64 + o]) = *(const uint4*)(other_g + g);
  }
  for (int idx = tid; idx < 2048; idx += 256) {
    int k = idx >> 4, n4 = (idx & 15) * 4;
    pack_b4(sW1P, k, n4, *(const float4*)(W1 + k * 64 + n4));
  }
  if (tid < 64) { sB1[tid] = b1[tid]; sW2[tid] = W2[tid]; }
  __syncthreads();

  const int wave = tid >> 5, lane = tid & 31;
  const int m0 = wave * 16;
  const int hl = lane >> 4, nn = lane & 15;

  // A fragments loaded once, reused across all n-tiles
  v16bf aC[4];
#pragma unroll
  for (int kt = 0; kt < 4; ++kt)
    aC[kt] = load_a_frag(&sCin[m0 * CIN_STR + kt * 32], CIN_STR, lane);

  float qacc[8];
#pragma unroll
  for (int v = 0; v < 8; ++v) qacc[v] = 0.f;

#pragma unroll
  for (int n0 = 0; n0 < 64; n0 += 16) {
    v8f c = {0.f, 0.f, 0.f, 0.f, 0.f, 0.f, 0.f, 0.f};
#pragma unroll
    for (int kt = 0; kt < 4; ++kt) {
      v16bf bf = load_b_packed(sW1P, kt, n0 >> 4, lane);
      c = __builtin_amdgcn_wmma_f32_16x16x32_bf16(false, aC[kt], false, bf, (short)0, c, false, false);
    }
    float bias = sB1[n0 + nn];
    float w2c  = sW2[n0 + nn];
#pragma unroll
    for (int v = 0; v < 8; ++v) {
      float x = c[v] + bias;
      x = x > 0.f ? x : 0.f;           // h = relu(...)
      qacc[v] += x * w2c;              // partial q contribution of this column
    }
  }

  // reduce each qacc[v] across the 16 lanes of this C-fragment half
#pragma unroll
  for (int v = 0; v < 8; ++v) {
    float s = qacc[v];
    s += __shfl_xor(s, 1, 32);
    s += __shfl_xor(s, 2, 32);
    s += __shfl_xor(s, 4, 32);
    s += __shfl_xor(s, 8, 32);
    qacc[v] = s;
  }
  float b2v = b2[0];
  if (nn == 0) {
#pragma unroll
    for (int v = 0; v < 8; ++v) {
      int row = m0 + v + 8 * hl;
      out[(size_t)ag * BATCH + b0 + row] = qacc[v] + b2v;
    }
  }
}

extern "C" void kernel_launch(void* const* d_in, const int* in_sizes, int n_in,
                              void* d_out, int out_size, void* d_ws, size_t ws_size,
                              hipStream_t stream) {
  const float* states  = (const float*)d_in[0];
  const float* actions = (const float*)d_in[1];
  const float* Ws   = (const float*)d_in[2];
  const float* bs   = (const float*)d_in[3];
  const float* We   = (const float*)d_in[4];
  const float* be   = (const float*)d_in[5];
  const float* Wk   = (const float*)d_in[6];
  const float* Wsel = (const float*)d_in[7];
  const float* Wv   = (const float*)d_in[8];
  const float* bv   = (const float*)d_in[9];
  const float* W1   = (const float*)d_in[10];
  const float* b1   = (const float*)d_in[11];
  const float* W2   = (const float*)d_in[12];
  const float* b2   = (const float*)d_in[13];
  float* out = (float*)d_out;

  const size_t N = (size_t)A_AGENTS * BATCH * HDIM;  // 16.8M elems per bf16 array
  __bf16* saenc = (__bf16*)d_ws;
  __bf16* keys  = saenc + N;
  __bf16* vals  = keys + N;
  __bf16* sel   = vals + N;
  __bf16* other = sel + N;

  dim3 blk(256);
  k_encode<<<dim3(A_AGENTS * (BATCH / 128)), blk, 0, stream>>>(
      states, actions, Ws, bs, We, be, Wk, Wsel, Wv, bv, saenc, keys, vals, sel);
  k_attend<<<dim3(BATCH / 32), blk, 0, stream>>>(keys, sel, vals, other);
  k_critic<<<dim3(A_AGENTS * (BATCH / 128)), blk, 0, stream>>>(
      saenc, other, W1, b1, W2, b2, out);
}
